// CausalSelfAttention_31542239822259
// MI455X (gfx1250) — matrix operand order repaired and checked by reading, not compiled, not run on hardware
//
#include <hip/hip_runtime.h>
#include <hip/hip_bf16.h>

// Problem constants (match reference)
#define BB   2
#define SS   2048
#define HIDD 1024
#define NHH  16
#define HDD  64
#define OPSZ 1024              // NH*HD
#define NCOL 3072              // Q 1024 | K 1024 | V 1024
#define ROWS 4096              // B*S

typedef __attribute__((ext_vector_type(16))) __bf16 v16bf;
typedef __attribute__((ext_vector_type(8)))  __bf16 v8bf;
typedef __attribute__((ext_vector_type(8)))  float  v8f;
typedef __attribute__((ext_vector_type(4)))  float  v4f;

static __device__ inline v8f wmma_bf16(v16bf a, v16bf b, v8f c) {
  // D = A(16x32 bf16) * B(32x16 bf16) + C(16x16 f32)
  return __builtin_amdgcn_wmma_f32_16x16x32_bf16(
      false, a, false, b, (short)0, c, false, false);
}

// ---- CDNA5 async global->LDS copy (ASYNCcnt-tracked), 16 bytes per lane ----
static __device__ inline void async_ld16(const __bf16* gsrc, __bf16* ldst) {
  unsigned loff = (unsigned)(size_t)ldst;   // low 32 bits of flat addr = LDS offset
  asm volatile("global_load_async_to_lds_b128 %0, %1, off"
               :: "v"(loff), "v"(gsrc) : "memory");
}
static __device__ inline void wait_async_all()  { asm volatile("s_wait_asynccnt 0x0"  ::: "memory"); }
static __device__ inline void wait_async_prev() { asm volatile("s_wait_asynccnt 0x10" ::: "memory"); }
static __device__ inline void wait_ds_all()     { asm volatile("s_wait_dscnt 0x0"     ::: "memory"); }

// ---------------------------------------------------------------------------
// Prep kernel 1: hidden_states fp32 -> bf16 (coalesced, 8 elems/thread)
// ---------------------------------------------------------------------------
__global__ void __launch_bounds__(256)
cvt_hs_bf16(const float* __restrict__ hs, __bf16* __restrict__ hsb)
{
  const size_t i = ((size_t)blockIdx.x * 256 + threadIdx.x) * 8;
  v4f a = *(const v4f*)(hs + i);
  v4f b = *(const v4f*)(hs + i + 4);
  v8bf o;
#pragma unroll
  for (int j = 0; j < 4; ++j) { o[j] = (__bf16)a[j]; o[4 + j] = (__bf16)b[j]; }
  *(v8bf*)(hsb + i) = o;
}

// ---------------------------------------------------------------------------
// Prep kernel 2: W = [Wqk | Wv] fp32 [1024 x 3072] -> bf16 transposed
// Wt[n][k], via 32x32 LDS tiles so both global sides are coalesced.
// ---------------------------------------------------------------------------
__global__ void __launch_bounds__(256)
wt_transpose(const float* __restrict__ Wqk, const float* __restrict__ Wv,
             __bf16* __restrict__ Wt)
{
  __shared__ float tile[32][33];
  const int tx = threadIdx.x & 31;
  const int ty = threadIdx.x >> 5;            // 0..7
  const int NTI = NCOL / 32;                  // 96
  const int n0 = (blockIdx.x % NTI) * 32;
  const int k0 = (blockIdx.x / NTI) * 32;

#pragma unroll
  for (int i = 0; i < 4; ++i) {
    const int k = k0 + ty + i * 8;
    const int n = n0 + tx;                    // 32-tiles never straddle col 2048
    tile[ty + i * 8][tx] = (n < 2 * OPSZ) ? Wqk[(size_t)k * (2 * OPSZ) + n]
                                          : Wv[(size_t)k * OPSZ + (n - 2 * OPSZ)];
  }
  __syncthreads();
#pragma unroll
  for (int i = 0; i < 4; ++i) {
    const int n = n0 + ty + i * 8;
    Wt[(size_t)n * HIDD + k0 + tx] = (__bf16)tile[tx][ty + i * 8];
  }
}

// ---------------------------------------------------------------------------
// Kernel 3: fused QKV projection GEMM.  Each wave computes a 16x64 tile of
// C[4096,3072] = hsb @ Wt^T + bias with 4 bf16 WMMAs per 32-deep K-step
// (1 contiguous 32B A-load + 4 contiguous 32B B-loads per lane per step).
// Q,K written bf16 [b,h,s,d]; V written transposed bf16 [b,h,d,s].
// ---------------------------------------------------------------------------
__global__ void __launch_bounds__(128)
qkv_proj_wmma(const __bf16* __restrict__ hsb, const __bf16* __restrict__ Wt,
              const float* __restrict__ bqk, const float* __restrict__ bv,
              __bf16* __restrict__ Qb, __bf16* __restrict__ Kb,
              __bf16* __restrict__ Vt)
{
  const int lane = threadIdx.x & 31;
  const int wave = threadIdx.x >> 5;
  const int g    = lane >> 4;
  const int ln   = lane & 15;

  const int NT   = NCOL / 64;                 // 48 col tiles
  const int tile = blockIdx.x * 4 + wave;     // [0, 256*48)
  const int mbase = (tile / NT) * 16;
  const int nbase = (tile % NT) * 64;

  const __bf16* arow = hsb + (size_t)(mbase + ln) * HIDD;

  v8f acc[4];
#pragma unroll
  for (int c = 0; c < 4; ++c)
#pragma unroll
    for (int i = 0; i < 8; ++i) acc[c][i] = 0.f;

  for (int kk = 0; kk < HIDD; kk += 32) {
    v16bf af;
    {
      v8bf lo = *(const v8bf*)(arow + kk + g * 16);
      v8bf hi = *(const v8bf*)(arow + kk + g * 16 + 8);
#pragma unroll
      for (int i = 0; i < 8; ++i) { af[i] = lo[i]; af[8 + i] = hi[i]; }
    }
#pragma unroll
    for (int c = 0; c < 4; ++c) {
      const __bf16* brow = Wt + (size_t)(nbase + c * 16 + ln) * HIDD + kk + g * 16;
      v16bf bf = *(const v16bf*)brow;
      acc[c] = wmma_bf16(af, bf, acc[c]);
    }
  }

  // Region (Q/K/V) is uniform across the wave: nbase is 64-aligned, the
  // boundaries are at 1024/2048.  Force a scalar branch.
  const int rgn = __builtin_amdgcn_readfirstlane(nbase >> 10);   // 0,1,2

  if (rgn == 0) {                                   // ---- Q ----
#pragma unroll
    for (int c = 0; c < 4; ++c) {
      const int col  = nbase + c * 16 + ln;
      const float bias = bqk[col];
      const int h = col >> 6, d = col & 63;
#pragma unroll
      for (int r = 0; r < 8; ++r) {
        const int m = mbase + 8 * g + r;
        const int b = m >> 11, s = m & (SS - 1);
        Qb[(((size_t)(b * NHH + h)) * SS + s) * HDD + d] = (__bf16)(acc[c][r] + bias);
      }
    }
  } else if (rgn == 1) {                            // ---- K ----
#pragma unroll
    for (int c = 0; c < 4; ++c) {
      const int col  = nbase + c * 16 + ln;
      const float bias = bqk[col];
      const int ck = col - OPSZ;
      const int h = ck >> 6, d = ck & 63;
#pragma unroll
      for (int r = 0; r < 8; ++r) {
        const int m = mbase + 8 * g + r;
        const int b = m >> 11, s = m & (SS - 1);
        Kb[(((size_t)(b * NHH + h)) * SS + s) * HDD + d] = (__bf16)(acc[c][r] + bias);
      }
    }
  } else {                                          // ---- V (transposed) ----
#pragma unroll
    for (int c = 0; c < 4; ++c) {
      const int cv = nbase - 2 * OPSZ + c * 16 + ln;
      const float bias = bv[cv];
      const int h = cv >> 6, d = cv & 63;
#pragma unroll
      for (int r = 0; r < 8; ++r) {
        const int m = mbase + 8 * g + r;
        const int b = m >> 11, s = m & (SS - 1);
        Vt[(((size_t)(b * NHH + h)) * HDD + d) * SS + s] = (__bf16)(acc[c][r] + bias);
      }
    }
  }
}

// ---------------------------------------------------------------------------
// Kernel 4: causal flash attention.  One wave owns a 16-row Q tile of one
// (b,h); streams 32-key blocks.  K (contiguous 4KB) and V^T tiles are
// double-buffered into LDS with global_load_async_to_lds_b128 (prefetch of
// block j+1 overlaps the 8 WMMAs + softmax of block j; s_wait_asynccnt 16
// fences only the buffer being consumed).
// ---------------------------------------------------------------------------
static __device__ inline void prefetch_kv(const __bf16* Kh, const __bf16* Vh,
                                          int kbase, __bf16* kdst, __bf16* vdst,
                                          int lane)
{
  // K tile: rows kbase..kbase+31 are contiguous 4KB in [b,h,s,d] layout
  const __bf16* kg = Kh + (size_t)kbase * HDD;
#pragma unroll
  for (int i = 0; i < 8; ++i) {
    const int off = i * 256 + lane * 8;       // bf16 elements (16B per lane)
    async_ld16(kg + off, kdst + off);
  }
  // V^T tile: 64 d-rows x 32 keys, global row stride S
  const int vr = lane >> 2;                   // 8 rows per instruction
  const int vc = (lane & 3) * 8;              // 4 lanes x 16B per row
#pragma unroll
  for (int i = 0; i < 8; ++i) {
    const int r = i * 8 + vr;
    async_ld16(Vh + (size_t)r * SS + kbase + vc, vdst + r * 32 + vc);
  }
}

__global__ void __launch_bounds__(64)
flash_attn_wmma(const __bf16* __restrict__ Qb, const __bf16* __restrict__ Kb,
                const __bf16* __restrict__ Vt, float* __restrict__ out)
{
  __shared__ __align__(128) __bf16 kbuf[2][2][32 * 64];   // [wave][buf]
  __shared__ __align__(128) __bf16 vbuf[2][2][64 * 32];
  __shared__ __align__(64)  __bf16 plds[2][16][32];

  const int lane = threadIdx.x & 31;
  const int wave = threadIdx.x >> 5;
  const int g    = lane >> 4;
  const int ln   = lane & 15;

  const int QT = SS / 16;                       // 128 q-tiles per (b,h)
  const int t  = blockIdx.x * 2 + wave;
  const int bh = t / QT;
  const int qbase = (t % QT) * 16;

  const __bf16* Qh = Qb + (size_t)bh * SS * HDD;
  const __bf16* Kh = Kb + (size_t)bh * SS * HDD;
  const __bf16* Vh = Vt + (size_t)bh * HDD * SS;

  // Q A-fragments (d-chunks 0..31, 32..63), resident in VGPRs for the loop
  v16bf qf[2];
  {
    const __bf16* qrow = Qh + (size_t)(qbase + ln) * HDD;
#pragma unroll
    for (int c = 0; c < 2; ++c) {
      v8bf lo = *(const v8bf*)(qrow + c * 32 + g * 16);
      v8bf hi = *(const v8bf*)(qrow + c * 32 + g * 16 + 8);
#pragma unroll
      for (int i = 0; i < 8; ++i) { qf[c][i] = lo[i]; qf[c][8 + i] = hi[i]; }
    }
  }

  v8f o[4];
  float mrow[8], lrow[8];
#pragma unroll
  for (int c = 0; c < 4; ++c)
#pragma unroll
    for (int i = 0; i < 8; ++i) o[c][i] = 0.f;
#pragma unroll
  for (int r = 0; r < 8; ++r) { mrow[r] = -3.0e38f; lrow[r] = 0.f; }

  const int nblocks = (qbase + 15) / 32 + 1;    // causal: keys <= qbase+15

  prefetch_kv(Kh, Vh, 0, kbuf[wave][0], vbuf[wave][0], lane);

  for (int j = 0; j < nblocks; ++j) {
    const int kbase = j * 32;
    if (j + 1 < nblocks) {
      prefetch_kv(Kh, Vh, kbase + 32, kbuf[wave][(j + 1) & 1],
                  vbuf[wave][(j + 1) & 1], lane);
      wait_async_prev();                        // buffer j resident
    } else {
      wait_async_all();
    }
    const __bf16* kc = kbuf[wave][j & 1];
    const __bf16* vc = vbuf[wave][j & 1];

    // ---- scores: two 16x16 key-half tiles, K-dim = d (2 x 32) ----
    v8f sc[2];
#pragma unroll
    for (int kh = 0; kh < 2; ++kh) {
      const __bf16* krow = kc + (kh * 16 + ln) * 64 + g * 16;
      v16bf b0 = *(const v16bf*)(krow);
      v16bf b1 = *(const v16bf*)(krow + 32);
      v8f s = {0.f, 0.f, 0.f, 0.f, 0.f, 0.f, 0.f, 0.f};
      s = wmma_bf16(qf[0], b0, s);
      s = wmma_bf16(qf[1], b1, s);
      sc[kh] = s;
    }

    // ---- scale, causal mask, online softmax ----
#pragma unroll
    for (int r = 0; r < 8; ++r) {
      const int qrow = qbase + 8 * g + r;
      float s0 = sc[0][r] * 0.125f;             // 1/sqrt(64)
      float s1 = sc[1][r] * 0.125f;
      if (kbase + ln > qrow)      s0 = -3.0e38f;
      if (kbase + 16 + ln > qrow) s1 = -3.0e38f;

      float mx = fmaxf(s0, s1);
#pragma unroll
      for (int off = 1; off < 16; off <<= 1)
        mx = fmaxf(mx, __shfl_xor(mx, off, 32));

      const float mnew  = fmaxf(mrow[r], mx);
      const float alpha = __expf(mrow[r] - mnew);
      const float p0 = __expf(s0 - mnew);
      const float p1 = __expf(s1 - mnew);

      float ps = p0 + p1;
#pragma unroll
      for (int off = 1; off < 16; off <<= 1)
        ps += __shfl_xor(ps, off, 32);

      lrow[r] = lrow[r] * alpha + ps;
      mrow[r] = mnew;
#pragma unroll
      for (int c = 0; c < 4; ++c) o[c][r] *= alpha;

      plds[wave][8 * g + r][ln]      = (__bf16)p0;
      plds[wave][8 * g + r][16 + ln] = (__bf16)p1;
    }

    wait_ds_all();                              // wave-private P round trip

    // ---- P as A-fragment (16 q rows x 32 keys) ----
    v16bf pf;
    {
      const __bf16* prow = &plds[wave][ln][0];
      v8bf lo = *(const v8bf*)(prow + g * 16);
      v8bf hi = *(const v8bf*)(prow + g * 16 + 8);
#pragma unroll
      for (int i = 0; i < 8; ++i) { pf[i] = lo[i]; pf[8 + i] = hi[i]; }
    }

    // ---- O += P V from LDS V^T ----
#pragma unroll
    for (int c = 0; c < 4; ++c) {
      v16bf vf = *(const v16bf*)(vc + (c * 16 + ln) * 32 + g * 16);
      o[c] = wmma_bf16(pf, vf, o[c]);
    }
  }

  // ---- normalize and write out[b, s, h*64 + d] ----
  const int b = bh / NHH, h = bh % NHH;
#pragma unroll
  for (int r = 0; r < 8; ++r) {
    const int qrow = qbase + 8 * g + r;
    const float inv = 1.0f / lrow[r];
    float* orow = out + ((size_t)b * SS + qrow) * OPSZ + h * HDD;
#pragma unroll
    for (int c = 0; c < 4; ++c)
      orow[c * 16 + ln] = o[c][r] * inv;
  }
}

// ---------------------------------------------------------------------------
extern "C" void kernel_launch(void* const* d_in, const int* in_sizes, int n_in,
                              void* d_out, int out_size, void* d_ws, size_t ws_size,
                              hipStream_t stream) {
  const float* hs  = (const float*)d_in[0];
  const float* Wqk = (const float*)d_in[1];
  const float* bqk = (const float*)d_in[2];
  const float* Wv  = (const float*)d_in[3];
  const float* bv  = (const float*)d_in[4];
  float* out = (float*)d_out;

  // Workspace (bf16): hsb 8MB | Wt 6MB | Q 8MB | K 8MB | V^T 8MB  = 38MB
  const size_t nHS = (size_t)ROWS * HIDD;     // 4,194,304
  const size_t nWT = (size_t)NCOL * HIDD;     // 3,145,728
  const size_t nQ  = (size_t)BB * NHH * SS * HDD;
  __bf16* hsb = (__bf16*)d_ws;
  __bf16* Wt  = hsb + nHS;
  __bf16* Qb  = Wt + nWT;
  __bf16* Kb  = Qb + nQ;
  __bf16* Vt  = Kb + nQ;

  cvt_hs_bf16<<<(int)(nHS / (256 * 8)), 256, 0, stream>>>(hs, hsb);           // 2048 blocks
  wt_transpose<<<(NCOL / 32) * (HIDD / 32), 256, 0, stream>>>(Wqk, Wv, Wt);   // 3072 blocks

  const int proj_blocks = (ROWS / 16) * (NCOL / 64) / 4;                      // 3072
  qkv_proj_wmma<<<proj_blocks, 128, 0, stream>>>(hsb, Wt, bqk, bv, Qb, Kb, Vt);

  const int attn_blocks = (BB * NHH * (SS / 16)) / 2;                         // 2048
  flash_attn_wmma<<<attn_blocks, 64, 0, stream>>>(Qb, Kb, Vt, out);
}
